// TimeLSTM_68066641707675
// MI455X (gfx1250) — compile-verified, matrix operand order checked by reading, not run on hardware
//
#include <hip/hip_runtime.h>
#include <hip/hip_bf16.h>
#include <stdint.h>

// ---------------- Problem constants ----------------
#define BATCH 64
#define SEQ   512
#define EMB   256
#define HID   512
#define FH    2048   // 4*HID
#define NWG   16     // persistent workgroups for the scan (one 32-col chunk of H each)

// Scan-kernel LDS staging: 4 gate blocks + W_d block, each 32 rows x 512 cols bf16
#define LDS_ROWS 160                       // 5 * 32
#define LDS_BYTES (LDS_ROWS * HID * 2)     // 163840 B  (< 320KB WGP LDS)

// ---------------- WMMA vector types (gfx1250, wave32) ----------------
typedef __attribute__((ext_vector_type(16))) __bf16 v16bf;
typedef __attribute__((ext_vector_type(8)))  __bf16 v8bf;
typedef __attribute__((ext_vector_type(8)))  float  v8f;

union FragU { v16bf v; v8bf h[2]; };

// float -> bf16 (round to nearest even)
__device__ __forceinline__ uint16_t f2bf(float f) {
    union { float f; uint32_t u; } v; v.f = f;
    uint32_t r = v.u + 0x7FFFu + ((v.u >> 16) & 1u);
    return (uint16_t)(r >> 16);
}
__device__ __forceinline__ float bf2f(uint16_t x) {
    union { uint32_t u; float f; } v; v.u = ((uint32_t)x) << 16;
    return v.f;
}
__device__ __forceinline__ float sigmoidf_(float x) {
    return 1.0f / (1.0f + __expf(-x));
}

// Load a 16x32 bf16 WMMA A/B fragment from row-major memory.
// ISA layout (CDNA5 ISA): lane l<16 -> row l, K = kb..kb+7 and kb+16..kb+23
// with kb = (lane>>4)*8.  Works for both A (row = activation row) and
// B (row = weight row, since B[k][n] = W[n][k] for D = act @ W^T).
// Force-inlined so clang's address-space inference emits ds_load_b128 when
// the base pointer is LDS and global_load_b128 when it is global.
__device__ __forceinline__ v16bf load_frag(const uint16_t* __restrict__ base,
                                           int ld, int row_base, int k0, int lane) {
    const int r  = lane & 15;
    const int kb = (lane >> 4) << 3;
    const uint16_t* p = base + (size_t)(row_base + r) * ld + k0 + kb;
    FragU f;
    f.h[0] = *(const v8bf*)(p);
    f.h[1] = *(const v8bf*)(p + 16);
    return f.v;
}

__device__ __forceinline__ v8f wmma_bf16(v16bf a, v16bf b, v8f c) {
    return __builtin_amdgcn_wmma_f32_16x16x32_bf16(
        /*neg_a=*/false, a, /*neg_b=*/false, b,
        /*c_mod=*/(short)0, c, /*reuse_a=*/false, /*reuse_b=*/false);
}

// ---------------- Prep kernels ----------------
__global__ void cvt_bf16_kernel(const float* __restrict__ src,
                                uint16_t* __restrict__ dst, long n) {
    long i = (long)blockIdx.x * blockDim.x + threadIdx.x;
    long stride = (long)gridDim.x * blockDim.x;
    for (; i < n; i += stride) dst[i] = f2bf(src[i]);
}

__global__ void init_state_kernel(uint16_t* h_bf, uint16_t* c_bf, float* c_f32,
                                  unsigned* bar) {
    int i = blockIdx.x * blockDim.x + threadIdx.x;
    if (i < BATCH * HID) {          // zero buffer 0 of each double-buffered state
        h_bf[i] = 0; c_bf[i] = 0; c_f32[i] = 0.0f;
    }
    if (i == 0) *bar = 0u;
}

// ---------------- Phase 1: u_all = X @ U^T  (M=32768, N=2048, K=256) ----------------
// One 16x16 tile per wave, 8 waves/block. Output stored bf16.
__global__ __launch_bounds__(256)
void u_gemm_kernel(const uint16_t* __restrict__ Xbf,
                   const uint16_t* __restrict__ Ubf,
                   uint16_t* __restrict__ ubf) {
    const int lane = threadIdx.x & 31;
    const int tile = blockIdx.x * 8 + (threadIdx.x >> 5);
    const int mt = tile >> 7;      // 2048 row-tiles
    const int nt = tile & 127;     // 128 col-tiles
    const int mb = mt << 4, nb = nt << 4;

    v8f acc = {};
    #pragma unroll
    for (int k = 0; k < EMB; k += 32) {
        v16bf a = load_frag(Xbf, EMB, mb, k, lane);
        v16bf b = load_frag(Ubf, EMB, nb, k, lane);
        acc = wmma_bf16(a, b, acc);
    }
    const int row0 = mb + ((lane >> 4) << 3);
    const int col  = nb + (lane & 15);
    #pragma unroll
    for (int r = 0; r < 8; ++r)
        ubf[(size_t)(row0 + r) * FH + col] = f2bf(acc[r]);
}

// ---------------- Phase 2: persistent recurrent scan ----------------
__device__ __forceinline__ void grid_sync(unsigned* bar, unsigned target) {
    __threadfence();
    __syncthreads();
    if (threadIdx.x == 0) {
        atomicAdd(bar, 1u);
        while (*(volatile unsigned*)bar < target)
            __builtin_amdgcn_s_sleep(1);
    }
    __syncthreads();
    __threadfence();
}

__global__ __launch_bounds__(256)
void timelstm_scan_kernel(const uint16_t* __restrict__ Wall_bf,   // [4H, H] bf16
                          const uint16_t* __restrict__ Wd_bf,     // [H, H]  bf16
                          const float* __restrict__ Wall_b,       // [4H]
                          const float* __restrict__ Uall_b,       // [4H]
                          const float* __restrict__ Wd_b,         // [H]
                          const uint16_t* __restrict__ u_bf,      // [B*S, 4H] bf16
                          const float* __restrict__ ts,           // [B, S]
                          uint16_t* h_bf,                         // [2][B*H]
                          uint16_t* c_bf,                         // [2][B*H]
                          float* c_f32,                           // [2][B*H]
                          float* __restrict__ out,                // outputs|h|c
                          unsigned* bar) {
    extern __shared__ uint16_t smem[];    // [LDS_ROWS][HID] weight slice for this WG

    const int w    = blockIdx.x;          // column chunk: cols [32w, 32w+32)
    const int wave = threadIdx.x >> 5;    // 8 waves
    const int lane = threadIdx.x & 31;
    const int rb = (wave & 3) << 4;       // batch-row tile base (0/16/32/48)
    const int lc = (wave >> 2) << 4;      // local hidden-col tile base within WG (0/16)
    const int cb = w * 32 + lc;           // global hidden-col tile base
    const int col = cb + (lane & 15);     // this lane's hidden column

    // Fold both biases once (u_all in reference already carries U_all_b; we add it here)
    const float b_f = Wall_b[col]         + Uall_b[col];
    const float b_i = Wall_b[HID + col]   + Uall_b[HID + col];
    const float b_o = Wall_b[2*HID + col] + Uall_b[2*HID + col];
    const float b_g = Wall_b[3*HID + col] + Uall_b[3*HID + col];
    const float b_d = Wd_b[col];

    // --- One-time LDS staging of this WG's loop-invariant weight slice ---
    // LDS rows 0..127  : gate g block g*32 + j  <- Wall_bf row g*HID + 32w + j
    // LDS rows 128..159: W_d block       128+j  <- Wd_bf   row 32w + j
    {
        const int nvec = LDS_ROWS * (HID / 8);  // 16B vectors to copy
        for (int idx = threadIdx.x; idx < nvec; idx += 256) {
            const int row = idx >> 6;            // HID/8 = 64 vectors per row
            const int kv  = (idx & 63) << 3;     // element offset within row
            size_t grow;
            const uint16_t* src;
            if (row < 128) { src = Wall_bf; grow = (size_t)((row >> 5) * HID + 32 * w + (row & 31)); }
            else           { src = Wd_bf;   grow = (size_t)(32 * w + (row - 128)); }
            *(v8bf*)(&smem[(size_t)row * HID + kv]) =
                *(const v8bf*)(&src[grow * HID + kv]);
        }
    }
    __syncthreads();

    for (int s = 0; s < SEQ; ++s) {
        const uint16_t* hcur = h_bf + (size_t)(s & 1) * (BATCH * HID);
        const uint16_t* ccur = c_bf + (size_t)(s & 1) * (BATCH * HID);
        const float* cfcur   = c_f32 + (size_t)(s & 1) * (BATCH * HID);
        uint16_t* hnxt = h_bf + (size_t)((s + 1) & 1) * (BATCH * HID);
        uint16_t* cnxt = c_bf + (size_t)((s + 1) & 1) * (BATCH * HID);
        float* cfnxt   = c_f32 + (size_t)((s + 1) & 1) * (BATCH * HID);

        v8f accf = {}, acci = {}, acco = {}, accg = {}, accs = {};
        for (int k = 0; k < HID; k += 32) {
            v16bf ah = load_frag(hcur, HID, rb, k, lane);       // A: h rows (global/L2)
            v16bf ac = load_frag(ccur, HID, rb, k, lane);       // A: c rows (global/L2)
            v16bf bf_ = load_frag(smem, HID,   0 + lc, k, lane);  // B: gate f   (LDS)
            v16bf bi_ = load_frag(smem, HID,  32 + lc, k, lane);  // B: gate i   (LDS)
            v16bf bo_ = load_frag(smem, HID,  64 + lc, k, lane);  // B: gate o   (LDS)
            v16bf bg_ = load_frag(smem, HID,  96 + lc, k, lane);  // B: gate c~  (LDS)
            v16bf bd_ = load_frag(smem, HID, 128 + lc, k, lane);  // B: W_d      (LDS)
            accf = wmma_bf16(ah, bf_, accf);
            acci = wmma_bf16(ah, bi_, acci);
            acco = wmma_bf16(ah, bo_, acco);
            accg = wmma_bf16(ah, bg_, accg);
            accs = wmma_bf16(ac, bd_, accs);
        }

        // Wave-local epilogue: acc element r -> batch row rb + r + 8*(lane>>4), col = col
        #pragma unroll
        for (int r = 0; r < 8; ++r) {
            const int bb = rb + ((lane >> 4) << 3) + r;
            const float tt   = ts[(size_t)bb * SEQ + s];
            const float cold = cfcur[(size_t)bb * HID + col];
            const float cs1  = tanhf(accs[r] + b_d);
            const float cadj = (cold - cs1) + cs1 * tt;
            const size_t ub  = ((size_t)bb * SEQ + s) * FH;
            const float f = sigmoidf_(accf[r] + b_f + bf2f(u_bf[ub + col]));
            const float i = sigmoidf_(acci[r] + b_i + bf2f(u_bf[ub + HID + col]));
            const float o = sigmoidf_(acco[r] + b_o + bf2f(u_bf[ub + 2*HID + col]));
            const float g = sigmoidf_(accg[r] + b_g + bf2f(u_bf[ub + 3*HID + col]));
            const float cn = f * cadj + i * g;
            const float hn = o * tanhf(cn);
            out[((size_t)bb * SEQ + s) * HID + col] = hn;     // outputs[B,S,H]
            cfnxt[(size_t)bb * HID + col] = cn;
            cnxt[(size_t)bb * HID + col]  = f2bf(cn);
            hnxt[(size_t)bb * HID + col]  = f2bf(hn);
            if (s == SEQ - 1) {
                out[(size_t)BATCH*SEQ*HID + (size_t)bb*HID + col] = hn;             // final h
                out[(size_t)BATCH*SEQ*HID + (size_t)BATCH*HID + (size_t)bb*HID + col] = cn; // final c
            }
        }
        grid_sync(bar, (unsigned)(NWG * (s + 1)));
    }
}

// ---------------- Launcher ----------------
extern "C" void kernel_launch(void* const* d_in, const int* in_sizes, int n_in,
                              void* d_out, int out_size, void* d_ws, size_t ws_size,
                              hipStream_t stream) {
    (void)in_sizes; (void)n_in; (void)out_size; (void)ws_size;
    const float* inp   = (const float*)d_in[0];   // [B,S,E]
    const float* ts    = (const float*)d_in[1];   // [B,S]
    const float* Wall  = (const float*)d_in[2];   // [4H,H]
    const float* Wallb = (const float*)d_in[3];   // [4H]
    const float* Uall  = (const float*)d_in[4];   // [4H,E]
    const float* Uallb = (const float*)d_in[5];   // [4H]
    const float* Wd    = (const float*)d_in[6];   // [H,H]
    const float* Wdb   = (const float*)d_in[7];   // [H]
    float* out = (float*)d_out;
    char* ws = (char*)d_ws;

    // Workspace layout (256B aligned)
    size_t off = 0;
    auto take = [&](size_t bytes) { size_t o = off; off += (bytes + 255) & ~(size_t)255; return o; };
    uint16_t* Xbf    = (uint16_t*)(ws + take((size_t)BATCH*SEQ*EMB * 2));
    uint16_t* Wallbf = (uint16_t*)(ws + take((size_t)FH*HID * 2));
    uint16_t* Ubf    = (uint16_t*)(ws + take((size_t)FH*EMB * 2));
    uint16_t* Wdbf   = (uint16_t*)(ws + take((size_t)HID*HID * 2));
    uint16_t* ubf    = (uint16_t*)(ws + take((size_t)BATCH*SEQ*FH * 2));
    uint16_t* h_bf   = (uint16_t*)(ws + take((size_t)2*BATCH*HID * 2));
    uint16_t* c_bf   = (uint16_t*)(ws + take((size_t)2*BATCH*HID * 2));
    float*    c_f32  = (float*)   (ws + take((size_t)2*BATCH*HID * 4));
    unsigned* bar    = (unsigned*)(ws + take(256));

    // 1) Convert activations + weights to bf16
    cvt_bf16_kernel<<<2048, 256, 0, stream>>>(inp,  Xbf,    (long)BATCH*SEQ*EMB);
    cvt_bf16_kernel<<<512,  256, 0, stream>>>(Wall, Wallbf, (long)FH*HID);
    cvt_bf16_kernel<<<256,  256, 0, stream>>>(Uall, Ubf,    (long)FH*EMB);
    cvt_bf16_kernel<<<128,  256, 0, stream>>>(Wd,   Wdbf,   (long)HID*HID);
    init_state_kernel<<<(BATCH*HID + 255)/256, 256, 0, stream>>>(h_bf, c_bf, c_f32, bar);

    // 2) Full-GPU input projection: 262144 16x16 tiles, 8 waves/block
    u_gemm_kernel<<<32768, 256, 0, stream>>>(Xbf, Ubf, ubf);

    // 3) Persistent recurrent scan: 16 WGs, weights staged in 160KB LDS,
    //    one grid barrier per step
    timelstm_scan_kernel<<<NWG, 256, LDS_BYTES, stream>>>(Wallbf, Wdbf, Wallb, Uallb, Wdb,
                                                          ubf, ts, h_bf, c_bf, c_f32, out, bar);
}
